// CausalSelfAttention_82076825027289
// MI455X (gfx1250) — matrix operand order repaired and checked by reading, not compiled, not run on hardware
//
#include <hip/hip_runtime.h>

typedef __attribute__((ext_vector_type(16))) __bf16 v16bf;
typedef __attribute__((ext_vector_type(8)))  float  v8f;
typedef unsigned short u16;
typedef unsigned int   u32;

#define LOG2E 1.44269504088896340736f

__device__ __forceinline__ u16 f2bf(float f) {
  u32 u = __builtin_bit_cast(u32, f);
  u32 r = (u + 0x7FFFu + ((u >> 16) & 1u)) >> 16;   // round-to-nearest-even
  return (u16)r;
}

// Build a 16xbf16 WMMA fragment from two 16-byte LDS chunks (ds_load_b128 x2).
__device__ __forceinline__ v16bf make_frag(const u16* p0, const u16* p1) {
  union { struct { uint4 a; uint4 b; } s; v16bf v; } u;
  u.s.a = *(const uint4*)p0;
  u.s.b = *(const uint4*)p1;
  return u.v;
}

__device__ __forceinline__ v8f wmma_bf16(v16bf a, v16bf b, v8f c) {
  return __builtin_amdgcn_wmma_f32_16x16x32_bf16(
      /*neg_a=*/false, a, /*neg_b=*/false, b,
      /*c_mod=*/(short)0, c, /*reuse_a=*/false, /*reuse_b=*/false);
}

// ---- CDNA5 async global->LDS copy (ASYNCcnt-tracked, no VGPR round trip) ----
__device__ __forceinline__ unsigned lds_off(const void* p) {
  return (unsigned)(size_t)p;   // low 32 bits of generic ptr == LDS offset
}
__device__ __forceinline__ void async_b128(unsigned lds, const u16* base,
                                           unsigned byte_off) {
  asm volatile("global_load_async_to_lds_b128 %0, %1, %2"
               :: "v"(lds), "v"(byte_off), "s"(base) : "memory");
}
__device__ __forceinline__ void wait_async0() {
  asm volatile("s_wait_asynccnt 0x0" ::: "memory");
}

// ---------------------------------------------------------------- convert
__global__ __launch_bounds__(256) void cvt_f32_bf16(
    const float* __restrict__ in, u16* __restrict__ out, int n) {
  int i = blockIdx.x * 256 + threadIdx.x;
  int stride = gridDim.x * 256;
  for (; i < n; i += stride) out[i] = f2bf(in[i]);
}

// in: [R][Cc] f32 row-major -> out: [Cc][R] bf16 row-major (LDS-tiled 64x64)
__global__ __launch_bounds__(128) void cvt_transpose(
    const float* __restrict__ in, u16* __restrict__ out, int R, int Cc) {
  __shared__ u16 tile[64 * 72];
  const int tid = threadIdx.x;
  const int r0 = blockIdx.y * 64, c0 = blockIdx.x * 64;
  #pragma unroll
  for (int it = 0; it < 8; ++it) {
    int chunk = tid + it * 128;          // 1024 chunks of 4 floats
    int r = chunk >> 4, cc = (chunk & 15) * 4;
    float4 f = *(const float4*)&in[(size_t)(r0 + r) * Cc + c0 + cc];
    u16* p = &tile[r * 72 + cc];
    p[0] = f2bf(f.x); p[1] = f2bf(f.y); p[2] = f2bf(f.z); p[3] = f2bf(f.w);
  }
  __syncthreads();
  #pragma unroll
  for (int it = 0; it < 4; ++it) {
    int chunk = tid + it * 128;          // 512 chunks of 8 bf16
    int c = chunk >> 3, rc = (chunk & 7) * 8;
    union { uint4 q; u16 h[8]; } u;
    #pragma unroll
    for (int e = 0; e < 8; ++e) u.h[e] = tile[(rc + e) * 72 + c];
    *(uint4*)&out[(size_t)(c0 + c) * R + r0 + rc] = u.q;
  }
}

// v [B,H,T,D] bf16 -> vt [B,H,D,T] bf16 (LDS-tiled 64x64)
__global__ __launch_bounds__(128) void transpose_v(
    const u16* __restrict__ v, u16* __restrict__ vt) {
  __shared__ u16 tile[64 * 72];
  const int tid = threadIdx.x;
  const int bh = blockIdx.y;
  const int t0 = blockIdx.x * 64;
  const size_t ibase = ((size_t)bh * 2048 + t0) * 64;
  #pragma unroll
  for (int it = 0; it < 4; ++it) {
    int chunk = tid + it * 128;
    int t = chunk >> 3, dc = (chunk & 7) * 8;
    *(uint4*)&tile[t * 72 + dc] = *(const uint4*)&v[ibase + (size_t)t * 64 + dc];
  }
  __syncthreads();
  const size_t obase = (size_t)bh * 64 * 2048 + t0;
  #pragma unroll
  for (int it = 0; it < 4; ++it) {
    int chunk = tid + it * 128;
    int d = chunk >> 3, tc = (chunk & 7) * 8;
    union { uint4 q; u16 h[8]; } u;
    #pragma unroll
    for (int e = 0; e < 8; ++e) u.h[e] = tile[(tc + e) * 72 + d];
    *(uint4*)&vt[obase + (size_t)d * 2048 + tc] = u.q;
  }
}

// ---------------------------------------------------------------- GEMM core
// C-tile 128x128, 8 waves, wave = 4x2 WMMA tiles. A [M][K], Bt [N][K] bf16.
// Double-buffered LDS, filled with global_load_async_to_lds_b128.
__device__ __forceinline__ void gemm_prefetch(
    int tid, const u16* __restrict__ A, const u16* __restrict__ Bt,
    int K, int m0, int n0, int kc, u16* bufA, u16* bufB) {
  #pragma unroll
  for (int it = 0; it < 2; ++it) {
    int chunk = tid + it * 256;          // 512 chunks of 8 bf16
    int row = chunk >> 2;
    int col = (chunk & 3) * 8;
    async_b128(lds_off(&bufA[row * 40 + col]), A,
               (unsigned)(((m0 + row) * K + kc + col) * 2));
    async_b128(lds_off(&bufB[row * 40 + col]), Bt,
               (unsigned)(((n0 + row) * K + kc + col) * 2));
  }
}

// One 32-deep WMMA step; cA/cB must be compile-time-traceable LDS pointers
// so the fragment loads lower to ds_load_b128 (not flat_load).
__device__ __forceinline__ void gemm_step(
    const u16* cA, const u16* cB,
    int wm, int wn, int nn, int hi, v8f acc[4][2]) {
  v16bf bfr[2];
  #pragma unroll
  for (int nt = 0; nt < 2; ++nt) {
    const u16* p = &cB[(wn * 32 + nt * 16 + nn) * 40 + hi * 8];
    bfr[nt] = make_frag(p, p + 16);
  }
  #pragma unroll
  for (int mt = 0; mt < 4; ++mt) {
    const u16* p = &cA[(wm * 64 + mt * 16 + nn) * 40 + hi * 8];
    v16bf af = make_frag(p, p + 16);
    #pragma unroll
    for (int nt = 0; nt < 2; ++nt)
      acc[mt][nt] = wmma_bf16(af, bfr[nt], acc[mt][nt]);
  }
}

// Two-phase (even/odd) pipelined loop: buffer choice is structural, never a
// runtime-selected pointer, keeping all fragment loads on the DS path.
__device__ __forceinline__ void gemm_core(
    const u16* __restrict__ A, const u16* __restrict__ Bt,
    int K, int m0, int n0,
    u16* sA0, u16* sA1, u16* sB0, u16* sB1, v8f acc[4][2]) {
  const int tid  = threadIdx.x;
  const int lane = tid & 31;
  const int wid  = tid >> 5;
  const int wm = wid >> 2, wn = wid & 3;
  const int nn = lane & 15, hi = lane >> 4;
  const int nk = K / 32;
  gemm_prefetch(tid, A, Bt, K, m0, n0, 0, sA0, sB0);
  for (int i = 0; i < nk; i += 2) {
    wait_async0();
    __syncthreads();
    if (i + 1 < nk)
      gemm_prefetch(tid, A, Bt, K, m0, n0, (i + 1) * 32, sA1, sB1);
    gemm_step(sA0, sB0, wm, wn, nn, hi, acc);
    if (i + 1 >= nk) break;
    wait_async0();
    __syncthreads();
    if (i + 2 < nk)
      gemm_prefetch(tid, A, Bt, K, m0, n0, (i + 2) * 32, sA0, sB0);
    gemm_step(sA1, sB1, wm, wn, nn, hi, acc);
  }
}

// qkv = x @ w_attn (Bt pre-transposed), scattered to q/k/v in [B,H,T,D] bf16
__global__ __launch_bounds__(256) void gemm_qkv(
    const u16* __restrict__ xbf, const u16* __restrict__ watT,
    u16* __restrict__ qb, u16* __restrict__ kb, u16* __restrict__ vb) {
  __shared__ u16 sA[2][128 * 40];
  __shared__ u16 sB[2][128 * 40];
  const int m0 = blockIdx.y * 128, n0 = blockIdx.x * 128;
  v8f acc[4][2];
  const v8f z8 = {0.f,0.f,0.f,0.f,0.f,0.f,0.f,0.f};
  #pragma unroll
  for (int mt = 0; mt < 4; ++mt)
    #pragma unroll
    for (int nt = 0; nt < 2; ++nt) acc[mt][nt] = z8;
  gemm_core(xbf, watT, 1024, m0, n0, sA[0], sA[1], sB[0], sB[1], acc);
  const int lane = threadIdx.x & 31, wid = threadIdx.x >> 5;
  const int wm = wid >> 2, wn = wid & 3;
  const int nn = lane & 15, hi = lane >> 4;
  #pragma unroll
  for (int mt = 0; mt < 4; ++mt)
    #pragma unroll
    for (int nt = 0; nt < 2; ++nt)
      #pragma unroll
      for (int r = 0; r < 8; ++r) {
        int gm = m0 + wm * 64 + mt * 16 + r + 8 * hi;   // [0,8192)
        int gn = n0 + wn * 32 + nt * 16 + nn;           // [0,3072)
        int which = gn >> 10;
        int c = gn & 1023;
        int h = c >> 6, d = c & 63;
        int b = gm >> 11, t = gm & 2047;
        u16* dst = (which == 0) ? qb : (which == 1 ? kb : vb);
        dst[((size_t)(b * 16 + h) * 2048 + t) * 64 + d] = f2bf(acc[mt][nt][r]);
      }
}

// out = y @ w_proj (Bt pre-transposed), fp32 output
__global__ __launch_bounds__(256) void gemm_proj(
    const u16* __restrict__ ybf, const u16* __restrict__ wpT,
    float* __restrict__ out) {
  __shared__ u16 sA[2][128 * 40];
  __shared__ u16 sB[2][128 * 40];
  const int m0 = blockIdx.y * 128, n0 = blockIdx.x * 128;
  v8f acc[4][2];
  const v8f z8 = {0.f,0.f,0.f,0.f,0.f,0.f,0.f,0.f};
  #pragma unroll
  for (int mt = 0; mt < 4; ++mt)
    #pragma unroll
    for (int nt = 0; nt < 2; ++nt) acc[mt][nt] = z8;
  gemm_core(ybf, wpT, 1024, m0, n0, sA[0], sA[1], sB[0], sB[1], acc);
  const int lane = threadIdx.x & 31, wid = threadIdx.x >> 5;
  const int wm = wid >> 2, wn = wid & 3;
  const int nn = lane & 15, hi = lane >> 4;
  #pragma unroll
  for (int mt = 0; mt < 4; ++mt)
    #pragma unroll
    for (int nt = 0; nt < 2; ++nt)
      #pragma unroll
      for (int r = 0; r < 8; ++r) {
        int gm = m0 + wm * 64 + mt * 16 + r + 8 * hi;
        int gn = n0 + wn * 32 + nt * 16 + nn;
        out[(size_t)gm * 1024 + gn] = acc[mt][nt][r];
      }
}

// ---------------------------------------------------------------- attention
// Grid: (T/64, B*H). Block: 128 thr = 4 waves; wave owns 16 query rows.
// K tile [t][d] and Vt tile [d][t] are straight async copies (double-buffered).
__device__ __forceinline__ void attn_prefetch(
    int tid, const u16* __restrict__ kb, const u16* __restrict__ vbt,
    size_t baseK, size_t baseV, int kv0, u16* bK, u16* bV) {
  #pragma unroll
  for (int it = 0; it < 4; ++it) {
    int chunk = tid + it * 128;          // 512 chunks of 8 bf16
    int r = chunk >> 3;                  // t for K, d for V
    int c = (chunk & 7) * 8;             // d chunk for K, t chunk for V
    async_b128(lds_off(&bK[r * 72 + c]), kb,
               (unsigned)((baseK + (size_t)(kv0 + r) * 64 + c) * 2));
    async_b128(lds_off(&bV[r * 72 + c]), vbt,
               (unsigned)((baseV + (size_t)r * 2048 + kv0 + c) * 2));
  }
}

__global__ __launch_bounds__(128) void attn_fwd(
    const u16* __restrict__ qb, const u16* __restrict__ kb,
    const u16* __restrict__ vbt, u16* __restrict__ yb) {
  __shared__ u16 sK[2][64 * 72];   // [t_kv][d]  == Bt layout for Q@K^T
  __shared__ u16 sV[2][64 * 72];   // [d][t_kv]  == Bt layout for P@V
  __shared__ u16 sP[64 * 72];      // per-wave P staging [16 q][64 kv]
  const int tid = threadIdx.x, lane = tid & 31, wid = tid >> 5;
  const int nn = lane & 15, hi = lane >> 4;
  const int bh = blockIdx.y;
  const int q0 = blockIdx.x * 64;
  const size_t baseK = (size_t)bh * 2048 * 64;
  const size_t baseV = (size_t)bh * 64 * 2048;

  // Q A-fragments, d = [0,32) and [32,64), kept in registers
  v16bf aq[2];
  {
    const u16* qrow = qb + baseK + (size_t)(q0 + wid * 16 + nn) * 64;
    #pragma unroll
    for (int f = 0; f < 2; ++f) {
      const u16* p = qrow + f * 32 + hi * 8;
      aq[f] = make_frag(p, p + 16);
    }
  }
  const v8f z8 = {0.f,0.f,0.f,0.f,0.f,0.f,0.f,0.f};
  v8f o[4];
  #pragma unroll
  for (int j = 0; j < 4; ++j) o[j] = z8;
  float m_run[8], l_run[8];
  #pragma unroll
  for (int r = 0; r < 8; ++r) { m_run[r] = -3.0e38f; l_run[r] = 0.f; }

  const int ntiles = blockIdx.x + 1;   // causal: kv tiles 0..qtile
  attn_prefetch(tid, kb, vbt, baseK, baseV, 0, sK[0], sV[0]);
  for (int ti = 0; ti < ntiles; ++ti) {
    const int kv0 = ti * 64;
    const int cb = ti & 1;
    wait_async0();
    __syncthreads();
    if (ti + 1 < ntiles)
      attn_prefetch(tid, kb, vbt, baseK, baseV, (ti + 1) * 64,
                    sK[cb ^ 1], sV[cb ^ 1]);
    // S = Q @ K^T for this wave's 16 rows x 64 kv cols
    v8f s[4];
    #pragma unroll
    for (int j = 0; j < 4; ++j) {
      s[j] = z8;
      #pragma unroll
      for (int f = 0; f < 2; ++f) {
        const u16* p = &sK[cb][(j * 16 + nn) * 72 + f * 32 + hi * 8];
        s[j] = wmma_bf16(aq[f], make_frag(p, p + 16), s[j]);
      }
    }
    // scale (C^-0.5 = 1/32) + causal mask in C-layout
    #pragma unroll
    for (int j = 0; j < 4; ++j) {
      int col = kv0 + j * 16 + nn;
      #pragma unroll
      for (int r = 0; r < 8; ++r) {
        int row = q0 + wid * 16 + r + 8 * hi;
        float v = s[j][r] * 0.03125f;
        s[j][r] = (col > row) ? -3.0e38f : v;
      }
    }
    // online softmax: per-row reductions across the 16 lanes of a half-wave
    #pragma unroll
    for (int r = 0; r < 8; ++r) {
      float mx = fmaxf(fmaxf(s[0][r], s[1][r]), fmaxf(s[2][r], s[3][r]));
      #pragma unroll
      for (int msk = 8; msk >= 1; msk >>= 1)
        mx = fmaxf(mx, __shfl_xor(mx, msk, 32));
      float nm = fmaxf(m_run[r], mx);
      float alpha = exp2f((m_run[r] - nm) * LOG2E);
      m_run[r] = nm;
      float sum = 0.f;
      #pragma unroll
      for (int j = 0; j < 4; ++j) {
        float p = exp2f((s[j][r] - nm) * LOG2E);
        s[j][r] = p;
        sum += p;
      }
      #pragma unroll
      for (int msk = 8; msk >= 1; msk >>= 1)
        sum += __shfl_xor(sum, msk, 32);
      l_run[r] = l_run[r] * alpha + sum;
      #pragma unroll
      for (int j = 0; j < 4; ++j) o[j][r] *= alpha;
    }
    // stage P (bf16) through LDS: C-layout -> A-fragment layout
    #pragma unroll
    for (int j = 0; j < 4; ++j)
      #pragma unroll
      for (int r = 0; r < 8; ++r)
        sP[(wid * 16 + r + 8 * hi) * 72 + j * 16 + nn] = f2bf(s[j][r]);
    __syncthreads();
    // O += P @ V
    #pragma unroll
    for (int f = 0; f < 2; ++f) {
      const u16* pp = &sP[(wid * 16 + nn) * 72 + f * 32 + hi * 8];
      v16bf ap = make_frag(pp, pp + 16);
      #pragma unroll
      for (int j = 0; j < 4; ++j) {
        const u16* pv = &sV[cb][(j * 16 + nn) * 72 + f * 32 + hi * 8];
        o[j] = wmma_bf16(ap, make_frag(pv, pv + 16), o[j]);
      }
    }
  }
  // epilogue: normalize and scatter to y [B,T,C] bf16
  const int b = bh >> 4, h = bh & 15;
  #pragma unroll
  for (int r = 0; r < 8; ++r) {
    float inv = 1.0f / l_run[r];
    int t = q0 + wid * 16 + r + 8 * hi;
    #pragma unroll
    for (int j = 0; j < 4; ++j)
      yb[((size_t)(b * 2048 + t)) * 1024 + h * 64 + j * 16 + nn] =
          f2bf(o[j][r] * inv);
  }
}

// ---------------------------------------------------------------- launch
extern "C" void kernel_launch(void* const* d_in, const int* in_sizes, int n_in,
                              void* d_out, int out_size, void* d_ws, size_t ws_size,
                              hipStream_t stream) {
  (void)in_sizes; (void)n_in; (void)out_size; (void)ws_size;
  const float* x      = (const float*)d_in[0];
  const float* w_attn = (const float*)d_in[1];
  const float* w_proj = (const float*)d_in[2];
  float* out = (float*)d_out;

  const int Bb = 4, T = 2048, C = 1024;
  const size_t M = (size_t)Bb * T;           // 8192

  size_t off = 0;
  auto carve = [&](size_t bytes) -> void* {
    void* p = (char*)d_ws + off;
    off += (bytes + 255) & ~(size_t)255;
    return p;
  };
  u16* xbf  = (u16*)carve(M * C * 2);                 // x bf16 [M][C]
  u16* waT  = (u16*)carve((size_t)C * 3 * C * 2);     // w_attn^T bf16 [3C][C]
  u16* wpT  = (u16*)carve((size_t)C * C * 2);         // w_proj^T bf16 [C][C]
  u16* qb   = (u16*)carve(M * C * 2);                 // [B,H,T,D]
  u16* kb   = (u16*)carve(M * C * 2);                 // [B,H,T,D]
  u16* vb   = (u16*)carve(M * C * 2);                 // [B,H,T,D]
  u16* vbt  = (u16*)carve(M * C * 2);                 // [B,H,D,T]
  u16* ybf  = (u16*)carve(M * C * 2);                 // [B,T,C]

  cvt_f32_bf16<<<2048, 256, 0, stream>>>(x, xbf, (int)(M * C));
  cvt_transpose<<<dim3(3 * C / 64, C / 64), 128, 0, stream>>>(w_attn, waT, C, 3 * C);
  cvt_transpose<<<dim3(C / 64, C / 64), 128, 0, stream>>>(w_proj, wpT, C, C);

  gemm_qkv<<<dim3(3 * C / 128, M / 128), 256, 0, stream>>>(xbf, waT, qb, kb, vb);
  transpose_v<<<dim3(T / 64, Bb * 16), 128, 0, stream>>>(vb, vbt);
  attn_fwd<<<dim3(T / 64, Bb * 16), 128, 0, stream>>>(qb, kb, vbt, ybf);
  gemm_proj<<<dim3(C / 128, M / 128), 256, 0, stream>>>(ybf, wpT, out);
}
